// Encoder_21534966022568
// MI455X (gfx1250) — matrix operand order repaired and checked by reading, not compiled, not run on hardware
//
#include <hip/hip_runtime.h>

typedef __attribute__((ext_vector_type(16))) __bf16 v16bf;
typedef __attribute__((ext_vector_type(8)))  float  v8f;

// Native convert: lowers to v_cvt_pk_bf16_f32 on gfx1250 (verified in disasm).
__device__ __forceinline__ __bf16 cvt_bf16(float f) { return (__bf16)f; }

__global__ void zero_u32_kernel(unsigned int* __restrict__ p, long long n) {
  long long i = blockIdx.x * (long long)blockDim.x + threadIdx.x;
  if (i < n) p[i] = 0u;
}

__global__ void deg_kernel(const int* __restrict__ dst, unsigned int* __restrict__ deg, int E) {
  int e = blockIdx.x * blockDim.x + threadIdx.x;
  if (e < E)
    __hip_atomic_fetch_add(&deg[dst[e]], 1u, __ATOMIC_RELAXED, __HIP_MEMORY_SCOPE_AGENT);
}

__global__ void dinv_kernel(const unsigned int* __restrict__ deg, float* __restrict__ dinv, int N) {
  int i = blockIdx.x * blockDim.x + threadIdx.x;
  if (i < N) dinv[i] = rsqrtf((float)deg[i] + 1.0f);
}

// Pack fp32 weight W[K,Nn] (row-major) into bf16 WMMA B-fragments:
// P[((kb*ntiles + ntile)*32 + lane)*16 + i];  B-layout: n = lane&15,
// k = kb*32 + 2*(i/2) + 16*(lane>>4) + (i&1)
__global__ void packB_kernel(const float* __restrict__ W, __bf16* __restrict__ P,
                             int K, int Nn) {
  int t = blockIdx.x * blockDim.x + threadIdx.x;
  int ntiles = Nn >> 4, kbt = K >> 5;
  int total = kbt * ntiles * 512;
  if (t >= total) return;
  int i = t & 15, lane = (t >> 4) & 31, rem = t >> 9;
  int ntile = rem % ntiles, kb = rem / ntiles;
  int v = i >> 1, lo = i & 1, half = lane >> 4;
  int n = ntile * 16 + (lane & 15);
  int k = kb * 32 + 2 * v + half * 16 + lo;
  P[t] = cvt_bf16(W[k * Nn + n]);
}

// H[M, NT*16] = X[M, K] * W  (W pre-packed as bf16 B-fragments).
// One wave handles a 16-row M-tile across all NT column tiles; K = KB*32.
template <int NT, int KB>
__global__ void gemm_wmma_kernel(const float* __restrict__ X,
                                 const __bf16* __restrict__ PB,
                                 float* __restrict__ H, int M, int K) {
  int wave  = threadIdx.x >> 5;
  int lane  = threadIdx.x & 31;
  int mtile = blockIdx.x * (blockDim.x >> 5) + wave;
  if (mtile * 16 >= M) return;          // wave-uniform: EXEC stays all-ones for WMMA
  int half = lane >> 4;
  int r0   = lane & 15;
  const float* xrow = X + (size_t)(mtile * 16 + r0) * K;
  const v16bf* bp   = (const v16bf*)PB;

  v8f zero = {0.f, 0.f, 0.f, 0.f, 0.f, 0.f, 0.f, 0.f};
  v8f acc[NT];
#pragma unroll
  for (int nt = 0; nt < NT; ++nt) acc[nt] = zero;

#pragma unroll
  for (int kb = 0; kb < KB; ++kb) {
    // A-fragment: contiguous k order, same as packB (not the ISA table order):
    // vgpr v holds K = kb*32 + 2*v + 16*half .. +1 (K-pairs contiguous)
    v16bf a;
#pragma unroll
    for (int v = 0; v < 8; ++v) {
      int kk = kb * 32 + v * 2 + half * 16;
      float2 f2 = *(const float2*)(xrow + kk);
      a[2 * v]     = cvt_bf16(f2.x);
      a[2 * v + 1] = cvt_bf16(f2.y);
    }
#pragma unroll
    for (int nt = 0; nt < NT; ++nt) {
      v16bf b = bp[(kb * NT + nt) * 32 + lane];
      acc[nt] = __builtin_amdgcn_wmma_f32_16x16x32_bf16(
          false, a, false, b, (short)0, acc[nt], false, false);
    }
  }
  // C/D layout: vgpr r -> M = r + 8*half, N = lane&15
  const int Nout = NT * 16;
#pragma unroll
  for (int nt = 0; nt < NT; ++nt) {
#pragma unroll
    for (int r = 0; r < 8; ++r) {
      H[(size_t)(mtile * 16 + half * 8 + r) * Nout + nt * 16 + r0] = acc[nt][r];
    }
  }
}

// AGG[dst[e], :] += H[src[e], :] * dinv[src]*dinv[dst]
// One thread per (edge, 4 features): coalesced b128 gather + 4 contiguous
// global_atomic_add_f32. F is a compile-time power of two -> shift/mask idx math.
template <int F>
__global__ void scatter_kernel(const float* __restrict__ H, const int* __restrict__ src,
                               const int* __restrict__ dst, const float* __restrict__ dinv,
                               float* AGG, int E) {
  constexpr int Q = F / 4;                 // float4 chunks per row
  long long idx = blockIdx.x * (long long)blockDim.x + threadIdx.x;
  long long total = (long long)E * Q;
  if (idx >= total) return;
  int e = (int)(idx / Q);
  int j = ((int)idx & (Q - 1)) * 4;
  int s = src[e], d = dst[e];
  float norm = dinv[s] * dinv[d];
  float4 h = *(const float4*)(H + (size_t)s * F + j);
  float* a = AGG + (size_t)d * F + j;
  __hip_atomic_fetch_add(a + 0, h.x * norm, __ATOMIC_RELAXED, __HIP_MEMORY_SCOPE_AGENT);
  __hip_atomic_fetch_add(a + 1, h.y * norm, __ATOMIC_RELAXED, __HIP_MEMORY_SCOPE_AGENT);
  __hip_atomic_fetch_add(a + 2, h.z * norm, __ATOMIC_RELAXED, __HIP_MEMORY_SCOPE_AGENT);
  __hip_atomic_fetch_add(a + 3, h.w * norm, __ATOMIC_RELAXED, __HIP_MEMORY_SCOPE_AGENT);
}

// In-place: AGGIO = AGGIO + H*dinv^2 + b  (optional ReLU), float4 per thread.
template <int F, bool RELU>
__global__ void finalize_kernel(const float* __restrict__ H, float* AGGIO,
                                const float* __restrict__ dinv, const float* __restrict__ b,
                                int N) {
  constexpr int Q = F / 4;
  long long idx = blockIdx.x * (long long)blockDim.x + threadIdx.x;
  long long total = (long long)N * Q;
  if (idx >= total) return;
  int i = (int)(idx / Q);
  int j = ((int)idx & (Q - 1)) * 4;
  float di = dinv[i];
  float d2 = di * di;
  float4 h  = *(const float4*)(H + (size_t)i * F + j);
  float4 ag = *(const float4*)(AGGIO + (size_t)i * F + j);
  float4 bb = *(const float4*)(b + j);
  float4 v;
  v.x = ag.x + h.x * d2 + bb.x;
  v.y = ag.y + h.y * d2 + bb.y;
  v.z = ag.z + h.z * d2 + bb.z;
  v.w = ag.w + h.w * d2 + bb.w;
  if (RELU) {
    v.x = fmaxf(v.x, 0.f); v.y = fmaxf(v.y, 0.f);
    v.z = fmaxf(v.z, 0.f); v.w = fmaxf(v.w, 0.f);
  }
  *(float4*)(AGGIO + (size_t)i * F + j) = v;
}

extern "C" void kernel_launch(void* const* d_in, const int* in_sizes, int n_in,
                              void* d_out, int out_size, void* d_ws, size_t ws_size,
                              hipStream_t stream) {
  const float* x  = (const float*)d_in[0];
  const float* W1 = (const float*)d_in[1];
  const float* b1 = (const float*)d_in[2];
  const float* W2 = (const float*)d_in[3];
  const float* b2 = (const float*)d_in[4];
  const int*   ei = (const int*)d_in[5];

  const int HID = in_sizes[2];               // 64
  const int OUT = in_sizes[4];               // 32
  const int IN  = in_sizes[1] / HID;         // 128
  const int N   = in_sizes[0] / IN;          // 50000
  const int E   = in_sizes[5] / 2;           // 800000
  const int* src = ei;
  const int* dst = ei + E;

  // Workspace layout (floats): deg[N] | dinv[N] | H1[64N] | AGG1[64N] | H2[32N] | P1 | P2
  float* w = (float*)d_ws;
  unsigned int* deg  = (unsigned int*)w;
  float*        dinv = w + (size_t)N;
  float*        H1   = w + (size_t)2 * N;
  float*        AGG1 = H1 + (size_t)N * HID;
  float*        H2   = AGG1 + (size_t)N * HID;
  __bf16*       P1   = (__bf16*)(H2 + (size_t)N * OUT);
  __bf16*       P2   = P1 + (size_t)(IN / 32) * (HID / 16) * 512;

  const int B = 256;
  auto blocks = [](long long n, int b) { return (unsigned)((n + b - 1) / b); };

  // 1) degree + dinv
  zero_u32_kernel<<<blocks(N, B), B, 0, stream>>>(deg, N);
  deg_kernel<<<blocks(E, B), B, 0, stream>>>(dst, deg, E);
  dinv_kernel<<<blocks(N, B), B, 0, stream>>>(deg, dinv, N);

  // 2) pack weights into bf16 WMMA B-fragments
  packB_kernel<<<blocks((IN / 32) * (HID / 16) * 512, B), B, 0, stream>>>(W1, P1, IN, HID);
  packB_kernel<<<blocks((HID / 32) * (OUT / 16) * 512, B), B, 0, stream>>>(W2, P2, HID, OUT);

  // 3) layer 1: H1 = X*W1 (WMMA), scatter-agg, self-loop+bias+ReLU in place
  int mtiles = (N + 15) / 16;
  gemm_wmma_kernel<4, 4><<<blocks(mtiles, 4), 128, 0, stream>>>(x, P1, H1, N, IN);
  zero_u32_kernel<<<blocks((long long)N * HID, B), B, 0, stream>>>((unsigned int*)AGG1, (long long)N * HID);
  scatter_kernel<64><<<blocks((long long)E * (HID / 4), B), B, 0, stream>>>(H1, src, dst, dinv, AGG1, E);
  finalize_kernel<64, true><<<blocks((long long)N * (HID / 4), B), B, 0, stream>>>(H1, AGG1, dinv, b1, N);

  // 4) layer 2: H2 = relu1*W2 (WMMA), scatter directly into d_out, finalize in place
  float* out = (float*)d_out;
  gemm_wmma_kernel<2, 2><<<blocks(mtiles, 4), 128, 0, stream>>>(AGG1, P2, H2, N, HID);
  zero_u32_kernel<<<blocks((long long)N * OUT, B), B, 0, stream>>>((unsigned int*)out, (long long)N * OUT);
  scatter_kernel<32><<<blocks((long long)E * (OUT / 4), B), B, 0, stream>>>(H2, src, dst, dinv, out, E);
  finalize_kernel<32, false><<<blocks((long long)N * (OUT / 4), B), B, 0, stream>>>(H2, out, dinv, b2, N);
}
